// MultiHeadAttention_21294447854258
// MI455X (gfx1250) — matrix operand order repaired and checked by reading, not compiled, not run on hardware
//
#include <hip/hip_runtime.h>

#define DMODEL 1024
#define SEQ    2048
#define NH     16
#define DKH    64
#define BATCH  2

typedef __attribute__((ext_vector_type(16))) __bf16 v16bf;
typedef __attribute__((ext_vector_type(8)))  float  v8f;
typedef unsigned int u32x4 __attribute__((ext_vector_type(4)));
typedef float        f32x4 __attribute__((ext_vector_type(4)));

union FragB {
  v16bf  v;
  u32x4  q[2];
  __bf16 b[16];
};

__device__ __forceinline__ __bf16 f2bf(float f) { return (__bf16)f; }  // HW cvt, RNE

__device__ __forceinline__ v8f wmma_bf16(v16bf a, v16bf b, v8f c) {
  // (neg_a, A, neg_b, B, c_mod, C, reuse_a, reuse_b)
  return __builtin_amdgcn_wmma_f32_16x16x32_bf16(false, a, false, b, (short)0, c,
                                                 false, false);
}

__device__ __forceinline__ v8f vzero8() {
  v8f z = {0.f, 0.f, 0.f, 0.f, 0.f, 0.f, 0.f, 0.f};
  return z;
}

// ---- Fragment loaders (ISA 7.12.2 layouts, wave32) ----------------------
// A (16x32, bf16): lanes 0-15 row M=lane, elems 0..7 -> K=hi*8+0..7,
// elems 8..15 -> K=16+hi*8+0..7 where hi = lane>>4.
__device__ __forceinline__ v16bf load_a_f32(const float* __restrict__ p, int ld,
                                            int m0, int k0, int lane) {
  const int l = lane & 15, hi = lane >> 4;
  const float* r = p + (size_t)(m0 + l) * ld + (k0 + hi * 8);
  FragB f;
#pragma unroll
  for (int e = 0; e < 8; ++e) {
    f.b[e]     = f2bf(r[e]);
    f.b[8 + e] = f2bf(r[16 + e]);
  }
  return f.v;
}

__device__ __forceinline__ v16bf load_a_bf16(const __bf16* __restrict__ p,
                                             int ld, int m0, int k0, int lane) {
  const int l = lane & 15, hi = lane >> 4;
  const __bf16* r = p + (size_t)(m0 + l) * ld + (k0 + hi * 8);
  const u32x4* v = (const u32x4*)r;   // 16B aligned chunks
  FragB f;
  f.q[0] = v[0];      // elems 0..7  (K = k0+hi*8 .. +7)
  f.q[1] = v[2];      // elems 8..15 (K = k0+16+hi*8 .. +7)
  return f.v;
}

// B (32x16, bf16): lane holds column N=lane&15; elems 0..7 -> K=hi*8+e, elems 8..15 -> K=16+hi*8+e.
// Source "row-major-in-K": element B[k][n] = src[n0+n][k0+k]
// (e.g. W[out,in], K^T, or transposed-V with k = sequence index).
__device__ __forceinline__ v16bf load_b_rowK_bf16(const __bf16* __restrict__ p,
                                                  int ld, int n0, int k0, int lane) {
  const int n = lane & 15, kh = lane >> 4;
  const __bf16* r = p + (size_t)(n0 + n) * ld + (k0 + kh * 8);
  const u32x4* v = (const u32x4*)r;
  FragB f;
  f.q[0] = v[0];
  f.q[1] = v[2];
  return f.v;
}

// ---- Kernel 0: elementwise f32 -> bf16 (weights pre-pass) ---------------
__global__ __launch_bounds__(256) void cvt_f32_to_bf16(const float* __restrict__ src,
                                                       __bf16* __restrict__ dst,
                                                       int n4) {
  const int i = blockIdx.x * blockDim.x + threadIdx.x;
  if (i >= n4) return;
  const f32x4 v = ((const f32x4*)src)[i];
  union { __bf16 b[4]; unsigned long long u; } o;
  o.b[0] = f2bf(v.x); o.b[1] = f2bf(v.y); o.b[2] = f2bf(v.z); o.b[3] = f2bf(v.w);
  ((unsigned long long*)dst)[i] = o.u;
}

// ---- Kernel 1: Y(bf16)[M,N] = X(f32)[M,K] @ Wb(bf16)[N,K]^T -------------
// Wave tile 32x64 (2x4 WMMA tiles); block = 8 waves -> 64x256.
__global__ __launch_bounds__(256) void gemm_xwT_f32_to_bf16(
    const float* __restrict__ X, const __bf16* __restrict__ Wb,
    __bf16* __restrict__ Y, int M, int N, int K) {
  const int lane = threadIdx.x & 31;
  const int wave = threadIdx.x >> 5;
  const int mtiles = M >> 6;
  const int mb = blockIdx.x % mtiles;
  const int nb = blockIdx.x / mtiles;
  const int m0 = mb * 64 + (wave & 1) * 32;
  const int n0 = nb * 256 + (wave >> 1) * 64;

  v8f acc[2][4];
#pragma unroll
  for (int i = 0; i < 2; ++i)
#pragma unroll
    for (int j = 0; j < 4; ++j) acc[i][j] = vzero8();

  for (int k0 = 0; k0 < K; k0 += 32) {
    v16bf a0 = load_a_f32(X, K, m0, k0, lane);
    v16bf a1 = load_a_f32(X, K, m0 + 16, k0, lane);
    v16bf b[4];
#pragma unroll
    for (int j = 0; j < 4; ++j) b[j] = load_b_rowK_bf16(Wb, K, n0 + j * 16, k0, lane);
#pragma unroll
    for (int j = 0; j < 4; ++j) {
      acc[0][j] = wmma_bf16(a0, b[j], acc[0][j]);
      acc[1][j] = wmma_bf16(a1, b[j], acc[1][j]);
    }
  }

  const int rh = lane >> 4, nc = lane & 15;
#pragma unroll
  for (int i = 0; i < 2; ++i)
#pragma unroll
    for (int j = 0; j < 4; ++j)
#pragma unroll
      for (int r = 0; r < 8; ++r) {
        const int row = m0 + i * 16 + rh * 8 + r;
        const int col = n0 + j * 16 + nc;
        Y[(size_t)row * N + col] = f2bf(acc[i][j][r]);
      }
}

// ---- Kernel 1T: same GEMM, epilogue writes Yt[b][col][seq] --------------
// (used for V so the attention P*V B-fragments become contiguous loads)
__global__ __launch_bounds__(256) void gemm_xwT_f32_to_bf16_T(
    const float* __restrict__ X, const __bf16* __restrict__ Wb,
    __bf16* __restrict__ Yt, int M, int N, int K) {
  const int lane = threadIdx.x & 31;
  const int wave = threadIdx.x >> 5;
  const int mtiles = M >> 6;
  const int mb = blockIdx.x % mtiles;
  const int nb = blockIdx.x / mtiles;
  const int m0 = mb * 64 + (wave & 1) * 32;
  const int n0 = nb * 256 + (wave >> 1) * 64;

  v8f acc[2][4];
#pragma unroll
  for (int i = 0; i < 2; ++i)
#pragma unroll
    for (int j = 0; j < 4; ++j) acc[i][j] = vzero8();

  for (int k0 = 0; k0 < K; k0 += 32) {
    v16bf a0 = load_a_f32(X, K, m0, k0, lane);
    v16bf a1 = load_a_f32(X, K, m0 + 16, k0, lane);
    v16bf b[4];
#pragma unroll
    for (int j = 0; j < 4; ++j) b[j] = load_b_rowK_bf16(Wb, K, n0 + j * 16, k0, lane);
#pragma unroll
    for (int j = 0; j < 4; ++j) {
      acc[0][j] = wmma_bf16(a0, b[j], acc[0][j]);
      acc[1][j] = wmma_bf16(a1, b[j], acc[1][j]);
    }
  }

  const int rh = lane >> 4, nc = lane & 15;
#pragma unroll
  for (int i = 0; i < 2; ++i)
#pragma unroll
    for (int j = 0; j < 4; ++j)
#pragma unroll
      for (int r = 0; r < 8; ++r) {
        const int row = m0 + i * 16 + rh * 8 + r;   // global token row
        const int col = n0 + j * 16 + nc;           // feature (d) index
        const int bbt = row >> 11;                  // row / SEQ
        const int s   = row & (SEQ - 1);
        Yt[(size_t)bbt * DMODEL * SEQ + (size_t)col * SEQ + s] = f2bf(acc[i][j][r]);
      }
}

// ---- Kernel 2: flash attention, one wave per (b, h, 16 query rows) ------
__global__ __launch_bounds__(256) void flash_attn_kernel(
    const __bf16* __restrict__ Qb, const __bf16* __restrict__ Kb,
    const __bf16* __restrict__ Vt, __bf16* __restrict__ Ctx) {
  __shared__ __align__(16) __bf16 Plds[8][16][32];

  const int lane = threadIdx.x & 31;
  const int wave = threadIdx.x >> 5;
  const int wid  = blockIdx.x * 8 + wave;
  const int qt   = wid & (SEQ / 16 - 1);      // 128 query tiles
  const int h    = (wid >> 7) & (NH - 1);
  const int bb   = wid >> 11;
  const int q0   = qt * 16;
  const size_t base  = (size_t)bb * SEQ * DMODEL + (size_t)h * DKH;
  const __bf16* VtB  = Vt + (size_t)bb * DMODEL * SEQ;   // [d_model][seq]
  const int rh = lane >> 4, nc = lane & 15;

  // Q fragments (16 x 64 over two K-steps), loaded once
  const v16bf qa0 = load_a_bf16(Qb + base, DMODEL, q0, 0, lane);
  const v16bf qa1 = load_a_bf16(Qb + base, DMODEL, q0, 32, lane);

  v8f accO[4];
#pragma unroll
  for (int d = 0; d < 4; ++d) accO[d] = vzero8();
  float mrun[8], lrun[8];
#pragma unroll
  for (int r = 0; r < 8; ++r) { mrun[r] = -3.0e38f; lrun[r] = 0.f; }

  const int jmax = (q0 + 15) >> 5;            // causal: last key block needed
  for (int j = 0; j <= jmax; ++j) {
    const int jb = j * 32;
    v8f s0 = vzero8(), s1 = vzero8();
    {
      v16bf k00 = load_b_rowK_bf16(Kb + base, DMODEL, jb, 0, lane);
      v16bf k01 = load_b_rowK_bf16(Kb + base, DMODEL, jb, 32, lane);
      v16bf k10 = load_b_rowK_bf16(Kb + base, DMODEL, jb + 16, 0, lane);
      v16bf k11 = load_b_rowK_bf16(Kb + base, DMODEL, jb + 16, 32, lane);
      s0 = wmma_bf16(qa0, k00, s0); s0 = wmma_bf16(qa1, k01, s0);
      s1 = wmma_bf16(qa0, k10, s1); s1 = wmma_bf16(qa1, k11, s1);
    }

#pragma unroll
    for (int r = 0; r < 8; ++r) {
      const int row = q0 + rh * 8 + r;
      float v0 = (jb + nc <= row)      ? s0[r] * 0.125f : -3.0e38f;
      float v1 = (jb + 16 + nc <= row) ? s1[r] * 0.125f : -3.0e38f;
      // row max across the 16 lanes of this half-wave
      float mx = fmaxf(v0, v1);
      mx = fmaxf(mx, __shfl_xor(mx, 1, 32));
      mx = fmaxf(mx, __shfl_xor(mx, 2, 32));
      mx = fmaxf(mx, __shfl_xor(mx, 4, 32));
      mx = fmaxf(mx, __shfl_xor(mx, 8, 32));
      const float mnew = fmaxf(mrun[r], mx);
      const float p0 = __expf(v0 - mnew);
      const float p1 = __expf(v1 - mnew);
      float rs = p0 + p1;
      rs += __shfl_xor(rs, 1, 32);
      rs += __shfl_xor(rs, 2, 32);
      rs += __shfl_xor(rs, 4, 32);
      rs += __shfl_xor(rs, 8, 32);
      const float alpha = __expf(mrun[r] - mnew);
      lrun[r] = lrun[r] * alpha + rs;
      mrun[r] = mnew;
#pragma unroll
      for (int d = 0; d < 4; ++d) accO[d][r] *= alpha;
      // stash P tile (C-layout -> row-major bf16) in wave-private LDS
      Plds[wave][rh * 8 + r][nc]      = f2bf(p0);
      Plds[wave][rh * 8 + r][16 + nc] = f2bf(p1);
    }

    // same-wave DS ordering: make the store->load explicit
    asm volatile("s_wait_dscnt 0" ::: "memory");
    const v16bf pa = load_a_bf16(&Plds[wave][0][0], 32, 0, 0, lane);
#pragma unroll
    for (int d = 0; d < 4; ++d) {
      // Vt is [d_model][seq]: B[k][n] = V[seq=jb+k][d=h*64+d*16+n]
      v16bf vb = load_b_rowK_bf16(VtB, SEQ, h * DKH + d * 16, jb, lane);
      accO[d] = wmma_bf16(pa, vb, accO[d]);
    }
  }

  // normalize and write ctx (bf16, [B,S,H*Dk] layout)
#pragma unroll
  for (int d = 0; d < 4; ++d)
#pragma unroll
    for (int r = 0; r < 8; ++r) {
      const int row = q0 + rh * 8 + r;
      const float o = accO[d][r] / lrun[r];
      Ctx[base + (size_t)row * DMODEL + d * 16 + nc] = f2bf(o);
    }
}

// ---- Kernel 3: out(f32)[M,N] = Ctx(bf16)[M,K] @ Wob(bf16)[N,K]^T --------
__global__ __launch_bounds__(256) void gemm_ctxwT_bf16_to_f32(
    const __bf16* __restrict__ A, const __bf16* __restrict__ Wb,
    float* __restrict__ Out, int M, int N, int K) {
  const int lane = threadIdx.x & 31;
  const int wave = threadIdx.x >> 5;
  const int mtiles = M >> 6;
  const int mb = blockIdx.x % mtiles;
  const int nb = blockIdx.x / mtiles;
  const int m0 = mb * 64 + (wave & 1) * 32;
  const int n0 = nb * 256 + (wave >> 1) * 64;

  v8f acc[2][4];
#pragma unroll
  for (int i = 0; i < 2; ++i)
#pragma unroll
    for (int j = 0; j < 4; ++j) acc[i][j] = vzero8();

  for (int k0 = 0; k0 < K; k0 += 32) {
    v16bf a0 = load_a_bf16(A, K, m0, k0, lane);
    v16bf a1 = load_a_bf16(A, K, m0 + 16, k0, lane);
    v16bf b[4];
#pragma unroll
    for (int j = 0; j < 4; ++j) b[j] = load_b_rowK_bf16(Wb, K, n0 + j * 16, k0, lane);
#pragma unroll
    for (int j = 0; j < 4; ++j) {
      acc[0][j] = wmma_bf16(a0, b[j], acc[0][j]);
      acc[1][j] = wmma_bf16(a1, b[j], acc[1][j]);
    }
  }

  const int rh = lane >> 4, nc = lane & 15;
#pragma unroll
  for (int i = 0; i < 2; ++i)
#pragma unroll
    for (int j = 0; j < 4; ++j)
#pragma unroll
      for (int r = 0; r < 8; ++r) {
        const int row = m0 + i * 16 + rh * 8 + r;
        const int col = n0 + j * 16 + nc;
        Out[(size_t)row * N + col] = acc[i][j][r];
      }
}

extern "C" void kernel_launch(void* const* d_in, const int* in_sizes, int n_in,
                              void* d_out, int out_size, void* d_ws, size_t ws_size,
                              hipStream_t stream) {
  const float* q  = (const float*)d_in[0];
  const float* k  = (const float*)d_in[1];
  const float* v  = (const float*)d_in[2];
  // d_in[3] is the causal mask (int32); causality is applied analytically.
  const float* wq = (const float*)d_in[4];
  const float* wk = (const float*)d_in[5];
  const float* wv = (const float*)d_in[6];
  const float* wo = (const float*)d_in[7];
  float* out = (float*)d_out;

  const int M = BATCH * SEQ;                    // 4096
  const int N = DMODEL, K = DMODEL;             // 1024
  const size_t elems = (size_t)M * DMODEL;      // 4 Mi elements
  const size_t welems = (size_t)DMODEL * DMODEL;// 1 Mi elements

  __bf16* Qb  = (__bf16*)d_ws;
  __bf16* Kb  = Qb + elems;
  __bf16* Vtb = Kb + elems;                     // V stored [B][DMODEL][SEQ]
  __bf16* Cx  = Vtb + elems;
  __bf16* Wqb = Cx + elems;
  __bf16* Wkb = Wqb + welems;
  __bf16* Wvb = Wkb + welems;
  __bf16* Wob = Wvb + welems;                   // total 40 MB bf16 workspace

  dim3 blk(256);

  // weight pre-conversion (once, vectorized)
  const int n4 = (int)(welems / 4);             // 262144
  dim3 cvt_grid((n4 + 255) / 256);
  cvt_f32_to_bf16<<<cvt_grid, blk, 0, stream>>>(wq, Wqb, n4);
  cvt_f32_to_bf16<<<cvt_grid, blk, 0, stream>>>(wk, Wkb, n4);
  cvt_f32_to_bf16<<<cvt_grid, blk, 0, stream>>>(wv, Wvb, n4);
  cvt_f32_to_bf16<<<cvt_grid, blk, 0, stream>>>(wo, Wob, n4);

  dim3 gemm_grid((M / 64) * (N / 256));         // 256 blocks
  gemm_xwT_f32_to_bf16  <<<gemm_grid, blk, 0, stream>>>(q, Wqb, Qb,  M, N, K);
  gemm_xwT_f32_to_bf16  <<<gemm_grid, blk, 0, stream>>>(k, Wkb, Kb,  M, N, K);
  gemm_xwT_f32_to_bf16_T<<<gemm_grid, blk, 0, stream>>>(v, Wvb, Vtb, M, N, K);

  dim3 fa_grid((BATCH * NH * (SEQ / 16)) / 8);  // 512 blocks, 8 waves each
  flash_attn_kernel<<<fa_grid, blk, 0, stream>>>(Qb, Kb, Vtb, Cx);

  gemm_ctxwT_bf16_to_f32<<<gemm_grid, blk, 0, stream>>>(Cx, Wob, out, M, N, K);
}
